// Encoder_33268816674929
// MI455X (gfx1250) — compile-verified, matrix operand order checked by reading, not run on hardware
//
#include <hip/hip_runtime.h>
#include <stdint.h>

// ---------------------------------------------------------------------------
// LSTM rollout for MI455X (gfx1250, wave32).
// h @ W_hh.T via V_WMMA_F32_16X16X32_BF16 (bf16 in, f32 accum), weights
// L2-resident (8.4 MB bf16 << 192 MB L2).
// - Gate rows of W_hh are permuted at pack time so each wave's 4 accumulator
//   tiles are (i,f,g,o) for the same 16 hidden units -> full LSTM cell update
//   in the GEMM epilogue, in registers. No gates buffer.
// - AR phase: the output projection is folded into the SAME kernel as a 5th
//   WMMA chain against a broadcast-packed W_out operand, so each step both
//   produces pred_{t-64} and consumes it as x -- one kernel per step on the
//   sequential critical path. h ping-pongs by step parity.
// ---------------------------------------------------------------------------

typedef __attribute__((ext_vector_type(16))) __bf16 v16bf;
typedef __attribute__((ext_vector_type(8)))  __bf16 v8bf;
typedef __attribute__((ext_vector_type(8)))  float  v8f;

#define HDIM   1024
#define G4H    4096            // 4*HDIM
#define NBATCH 512
#define TSEQ   64
#define NSTEPS 96
#define KT     (HDIM / 32)     // 32 k-steps of 32
#define BTILE_ELEMS (KT * 32 * 16)   // elems per packed n-tile column

__device__ __forceinline__ unsigned short f2bf(float f) {
  unsigned int u = __float_as_uint(f);
  u += 0x7FFFu + ((u >> 16) & 1u);          // round-to-nearest-even
  return (unsigned short)(u >> 16);
}
__device__ __forceinline__ float bf2f(unsigned short s) {
  return __uint_as_float(((unsigned int)s) << 16);
}
__device__ __forceinline__ float sigmoidf_(float x) {
  return 1.0f / (1.0f + __expf(-x));
}
__device__ __forceinline__ float tanhf_(float x) {
  float e = __expf(-2.0f * fabsf(x));       // stable for all x
  float t = (1.0f - e) / (1.0f + e);
  return copysignf(t, x);
}

// ---------------------------------------------------------------------------
// Pack W_hh [4096,1024] f32 -> bf16, lane-swizzled per the ISA 16-bit 16x32
// operand layout AND gate-interleaved:
//   packed n-tile nt -> gate = nt & 3, unit-group ug = nt >> 2
//   original row j = gate*HDIM + ug*16 + n,  n = lane&15.
// Lane's 16-element chunk element i -> k = (i<8 ? 8*half+i : 16+8*half+i-8).
// ---------------------------------------------------------------------------
__global__ __launch_bounds__(256) void pack_whh(
    const float* __restrict__ whh, unsigned short* __restrict__ wp) {
  int idx  = blockIdx.x * 256 + threadIdx.x;   // 0 .. 4096*1024-1
  int i    = idx & 15;
  int lane = (idx >> 4) & 31;
  int kt   = (idx >> 9) & 31;
  int nt   = idx >> 14;
  int half = lane >> 4;
  int n    = lane & 15;
  int k    = (i < 8) ? (8 * half + i) : (16 + 8 * half + (i - 8));
  int gate = nt & 3;                 // 0:i 1:f 2:g 3:o
  int ug   = nt >> 2;                // hidden-unit group (16 units)
  int row  = gate * HDIM + ug * 16 + n;
  int col  = kt * 32 + k;
  wp[idx] = f2bf(whh[row * HDIM + col]);
}

// Pack W_out [1024] as a broadcast B-operand: every column n of each 32x16
// k-tile equals the W_out slice -> D[m,n] = h[m,:].W_out for all n.
__global__ __launch_bounds__(256) void pack_wout(
    const float* __restrict__ wout, unsigned short* __restrict__ wop) {
  int idx  = blockIdx.x * 256 + threadIdx.x;   // 0 .. KT*32*16-1 (16384)
  int i    = idx & 15;
  int lane = (idx >> 4) & 31;
  int kt   = idx >> 9;
  int half = lane >> 4;
  int k    = (i < 8) ? (8 * half + i) : (16 + 8 * half + (i - 8));
  wop[idx] = f2bf(wout[kt * 32 + k]);
}

// h=0, c=0, combined bias bc = b_ih + b_hh
__global__ __launch_bounds__(256) void init_state(
    const float* __restrict__ bih, const float* __restrict__ bhh,
    unsigned short* __restrict__ hbf, float* __restrict__ c,
    float* __restrict__ bc) {
  int idx = blockIdx.x * 256 + threadIdx.x;    // 0 .. 512*1024-1
  hbf[idx] = 0;
  c[idx]   = 0.0f;
  if (idx < G4H) bc[idx] = bih[idx] + bhh[idx];
}

// ---------------------------------------------------------------------------
// One fused LSTM step. 2048 waves; wave = 16 batch rows x 16 hidden units x
// 4 gates. PRED=true (AR phase): a 5th WMMA chain computes
// pred = h_prev . W_out, used as x and written to out[b, sout].
// ---------------------------------------------------------------------------
template <bool PRED>
__global__ __launch_bounds__(256) void lstm_step(
    const unsigned short* __restrict__ hprev,
    const unsigned short* __restrict__ wp,
    const unsigned short* __restrict__ wop,
    unsigned short* __restrict__ hnext,
    float* __restrict__ c,
    const float* __restrict__ bc,  const float* __restrict__ wih,
    const float* __restrict__ data, const float* __restrict__ bout,
    float* __restrict__ out, int step, int sout) {
  const int wave = threadIdx.x >> 5;
  const int lane = threadIdx.x & 31;
  const int task = blockIdx.x * 8 + wave;      // 0..2047
  const int bm   = task >> 6;                  // 0..31  (batch tile)
  const int bn4  = task & 63;                  // 0..63  (hidden-unit group)
  const int m0   = bm * 16;
  const int half = lane >> 4;
  const int ulane = bn4 * 16 + (lane & 15);    // this lane's hidden unit

  // Per-lane gate constants: issue early, they land during the K-loop.
  const float bci = bc[ulane],            wii = wih[ulane];
  const float bcf = bc[HDIM + ulane],     wif = wih[HDIM + ulane];
  const float bcg = bc[2 * HDIM + ulane], wig = wih[2 * HDIM + ulane];
  const float bco = bc[3 * HDIM + ulane], wio = wih[3 * HDIM + ulane];
  const float bout0 = PRED ? bout[0] : 0.0f;

  const unsigned short* hrow  = hprev + (m0 + (lane & 15)) * HDIM + 8 * half;
  const unsigned short* bbase = wp + ((size_t)(bn4 * 4) * BTILE_ELEMS) + lane * 16;
  const unsigned short* wbase = wop + lane * 16;

  v8f acc0 = {}; v8f acc1 = {}; v8f acc2 = {}; v8f acc3 = {}; v8f acc4 = {};

  // ---- prologue: loads for kt = 0 ----
  v8bf  alo = *reinterpret_cast<const v8bf*>(hrow);
  v8bf  ahi = *reinterpret_cast<const v8bf*>(hrow + 16);
  v16bf b0  = *reinterpret_cast<const v16bf*>(bbase);
  v16bf b1  = *reinterpret_cast<const v16bf*>(bbase + 1 * BTILE_ELEMS);
  v16bf b2  = *reinterpret_cast<const v16bf*>(bbase + 2 * BTILE_ELEMS);
  v16bf b3  = *reinterpret_cast<const v16bf*>(bbase + 3 * BTILE_ELEMS);
  v16bf bw  = {};
  if constexpr (PRED) bw = *reinterpret_cast<const v16bf*>(wbase);

#pragma unroll 2
  for (int kt = 0; kt < KT - 1; ++kt) {
    // issue loads for kt+1 before consuming kt's operands
    const unsigned short* ha2 = hrow + (kt + 1) * 32;
    const unsigned short* bk2 = bbase + (kt + 1) * (32 * 16);
    v8bf  nalo = *reinterpret_cast<const v8bf*>(ha2);
    v8bf  nahi = *reinterpret_cast<const v8bf*>(ha2 + 16);
    v16bf nb0  = *reinterpret_cast<const v16bf*>(bk2);
    v16bf nb1  = *reinterpret_cast<const v16bf*>(bk2 + 1 * BTILE_ELEMS);
    v16bf nb2  = *reinterpret_cast<const v16bf*>(bk2 + 2 * BTILE_ELEMS);
    v16bf nb3  = *reinterpret_cast<const v16bf*>(bk2 + 3 * BTILE_ELEMS);
    v16bf nbw  = {};
    if constexpr (PRED)
      nbw = *reinterpret_cast<const v16bf*>(wbase + (kt + 1) * (32 * 16));

    v16bf A = __builtin_shufflevector(alo, ahi,
        0, 1, 2, 3, 4, 5, 6, 7, 8, 9, 10, 11, 12, 13, 14, 15);
    acc0 = __builtin_amdgcn_wmma_f32_16x16x32_bf16(false, A, false, b0, (short)0, acc0, false, false);
    acc1 = __builtin_amdgcn_wmma_f32_16x16x32_bf16(false, A, false, b1, (short)0, acc1, false, false);
    acc2 = __builtin_amdgcn_wmma_f32_16x16x32_bf16(false, A, false, b2, (short)0, acc2, false, false);
    acc3 = __builtin_amdgcn_wmma_f32_16x16x32_bf16(false, A, false, b3, (short)0, acc3, false, false);
    if constexpr (PRED)
      acc4 = __builtin_amdgcn_wmma_f32_16x16x32_bf16(false, A, false, bw, (short)0, acc4, false, false);

    alo = nalo; ahi = nahi;
    b0 = nb0; b1 = nb1; b2 = nb2; b3 = nb3;
    if constexpr (PRED) bw = nbw;
  }
  {
    v16bf A = __builtin_shufflevector(alo, ahi,
        0, 1, 2, 3, 4, 5, 6, 7, 8, 9, 10, 11, 12, 13, 14, 15);
    acc0 = __builtin_amdgcn_wmma_f32_16x16x32_bf16(false, A, false, b0, (short)0, acc0, false, false);
    acc1 = __builtin_amdgcn_wmma_f32_16x16x32_bf16(false, A, false, b1, (short)0, acc1, false, false);
    acc2 = __builtin_amdgcn_wmma_f32_16x16x32_bf16(false, A, false, b2, (short)0, acc2, false, false);
    acc3 = __builtin_amdgcn_wmma_f32_16x16x32_bf16(false, A, false, b3, (short)0, acc3, false, false);
    if constexpr (PRED)
      acc4 = __builtin_amdgcn_wmma_f32_16x16x32_bf16(false, A, false, bw, (short)0, acc4, false, false);
  }

  // ---- fused cell update. C/D layout: VGPR r, lane l ->
  //      batch b = m0 + 8*(l>>4) + r, unit u = ulane. acc0..3 = gi,gf,gg,go.
  //      acc4[r] = h_prev[b,:].W_out (same row mapping, all columns equal).
  const int brow0 = m0 + 8 * half;
#pragma unroll
  for (int r = 0; r < 8; ++r) {
    const int b = brow0 + r;
    float x;
    if constexpr (PRED) x = acc4[r] + bout0;        // pred_{sout} == x_t
    else                x = data[b * TSEQ + step];
    float gi = acc0[r] + bci + x * wii;
    float gf = acc1[r] + bcf + x * wif;
    float gg = acc2[r] + bcg + x * wig;
    float go = acc3[r] + bco + x * wio;
    float iv = sigmoidf_(gi);
    float fv = sigmoidf_(gf);
    float gv = tanhf_(gg);
    float ov = sigmoidf_(go);
    const int idx = b * HDIM + ulane;
    float cn = fv * c[idx] + iv * gv;
    c[idx] = cn;
    hnext[idx] = f2bf(ov * tanhf_(cn));
    if constexpr (PRED) {
      if (bn4 == 0 && (lane & 15) == 0)
        out[b * NSTEPS + sout] = x;                 // one writer per (b, sout)
    }
  }
}

// ---------------------------------------------------------------------------
// Tail: pred_95 = h_final . W_out + b_out  (one block per batch row).
// ---------------------------------------------------------------------------
__global__ __launch_bounds__(256) void lstm_pred_tail(
    const unsigned short* __restrict__ h, const float* __restrict__ wout,
    const float* __restrict__ bout, float* __restrict__ out) {
  __shared__ float red[256];
  const int b   = blockIdx.x;
  const int tid = threadIdx.x;
  float pw = 0.0f;
#pragma unroll
  for (int q = 0; q < 4; ++q) {
    int u = tid + q * 256;
    pw += bf2f(h[b * HDIM + u]) * wout[u];
  }
  red[tid] = pw;
  __syncthreads();
  for (int sft = 128; sft > 0; sft >>= 1) {
    if (tid < sft) red[tid] += red[tid + sft];
    __syncthreads();
  }
  if (tid == 0) out[b * NSTEPS + (NSTEPS - 1)] = red[0] + bout[0];
}

// ---------------------------------------------------------------------------
extern "C" void kernel_launch(void* const* d_in, const int* in_sizes, int n_in,
                              void* d_out, int out_size, void* d_ws, size_t ws_size,
                              hipStream_t stream) {
  const float* data = (const float*)d_in[0];   // [512,64,1]
  const float* wih  = (const float*)d_in[1];   // [4096,1]
  const float* whh  = (const float*)d_in[2];   // [4096,1024]
  const float* bih  = (const float*)d_in[3];   // [4096]
  const float* bhh  = (const float*)d_in[4];   // [4096]
  const float* wout = (const float*)d_in[5];   // [1,1024]
  const float* bout = (const float*)d_in[6];   // [1]
  float* out = (float*)d_out;                  // [512,96,1]

  char* ws = (char*)d_ws;
  size_t off = 0;
  unsigned short* wp  = (unsigned short*)(ws + off); off += (size_t)G4H * HDIM * 2;     // 8.4 MB
  unsigned short* wop = (unsigned short*)(ws + off); off += (size_t)KT * 32 * 16 * 2;   // 32 KB
  unsigned short* hA  = (unsigned short*)(ws + off); off += (size_t)NBATCH * HDIM * 2;  // 1 MB
  unsigned short* hB  = (unsigned short*)(ws + off); off += (size_t)NBATCH * HDIM * 2;  // 1 MB
  float* c            = (float*)(ws + off);          off += (size_t)NBATCH * HDIM * 4;  // 2 MB
  float* bc           = (float*)(ws + off);          off += (size_t)G4H * 4;

  hipLaunchKernelGGL(pack_whh, dim3((G4H * HDIM) / 256), dim3(256), 0, stream, whh, wp);
  hipLaunchKernelGGL(pack_wout, dim3((KT * 32 * 16) / 256), dim3(256), 0, stream, wout, wop);
  hipLaunchKernelGGL(init_state, dim3((NBATCH * HDIM) / 256), dim3(256), 0, stream,
                     bih, bhh, hA, c, bc);

  // 64 warmup steps + 95 autoregressive steps; one kernel per step.
  unsigned short* hlast = hA;
  for (int step = 0; step < TSEQ + NSTEPS - 1; ++step) {
    const unsigned short* hin = (step & 1) ? hB : hA;
    unsigned short*      hout = (step & 1) ? hA : hB;
    if (step < TSEQ) {
      hipLaunchKernelGGL((lstm_step<false>), dim3(256), dim3(256), 0, stream,
                         hin, wp, wop, hout, c, bc, wih, data, bout, out, step, 0);
    } else {
      hipLaunchKernelGGL((lstm_step<true>), dim3(256), dim3(256), 0, stream,
                         hin, wp, wop, hout, c, bc, wih, data, bout, out, step, step - TSEQ);
    }
    hlast = hout;
  }
  hipLaunchKernelGGL(lstm_pred_tail, dim3(NBATCH), dim3(256), 0, stream,
                     hlast, wout, bout, out);
}
